// TritonDecomposedAttentionBlock_78915729097147
// MI455X (gfx1250) — matrix operand order repaired and checked by reading, not compiled, not run on hardware
//
#include <hip/hip_runtime.h>

// ---------------------------------------------------------------------------
// Types for CDNA5 WMMA (wave32): v_wmma_f32_16x16x32_bf16
// ---------------------------------------------------------------------------
typedef __bf16 bf16_t;
typedef __attribute__((ext_vector_type(16))) bf16_t v16bf;
typedef __attribute__((ext_vector_type(8)))  bf16_t v8bf;
typedef __attribute__((ext_vector_type(8)))  float  v8f;
typedef __attribute__((ext_vector_type(4)))  int    v4i;

#define WMMA_BF16(A, B, C) \
  __builtin_amdgcn_wmma_f32_16x16x32_bf16(false, (A), false, (B), (short)0, (C), false, false)

static __device__ __forceinline__ unsigned short f2bf_bits(float f) {
  union { float f; unsigned u; } v; v.f = f;
  unsigned r = v.u + 0x7FFFu + ((v.u >> 16) & 1u);  // round-to-nearest-even
  return (unsigned short)(r >> 16);
}

// ---------------------------------------------------------------------------
// Async global->LDS copy (16 B per lane), tracked by ASYNCcnt.
// Builtin signature (from clang diagnostic): (int4 AS1*, int4 AS3*, int, int)
// ---------------------------------------------------------------------------
static __device__ __forceinline__ void async_copy16(bf16_t* lds_dst, const bf16_t* gsrc) {
#if __has_builtin(__builtin_amdgcn_global_load_async_to_lds_b128)
  __builtin_amdgcn_global_load_async_to_lds_b128(
      (__attribute__((address_space(1))) v4i*)gsrc,
      (__attribute__((address_space(3))) v4i*)lds_dst, 0, 0);
#else
  unsigned lofs = (unsigned)(unsigned long long)(__attribute__((address_space(3))) void*)lds_dst;
  asm volatile("global_load_async_to_lds_b128 %0, %1, off"
               :: "v"(lofs), "v"(gsrc) : "memory");
#endif
}

template <int N>
static __device__ __forceinline__ void wait_async() {
#if __has_builtin(__builtin_amdgcn_s_wait_asynccnt)
  __builtin_amdgcn_s_wait_asynccnt(N);
#else
  asm volatile("s_wait_asynccnt %0" :: "i"(N));
#endif
}

// ---------------------------------------------------------------------------
// Fragment loaders (ISA 7.12.2 layouts, wave32). Work for global or LDS ptrs.
// A (16x32 MxK) row-major, base at (row0,k0):
//   lane L: g=L>>4, r=L&15; elems 0..7 = K(g*8+0..7), 8..15 = K(16+g*8+0..7)
// ---------------------------------------------------------------------------
static __device__ __forceinline__ v16bf load_frag_a(const bf16_t* base, int ld, int lane) {
  const int g = lane >> 4, r = lane & 15;
  const bf16_t* p = base + (size_t)r * ld + g * 8;
  v8bf lo = *(const v8bf*)(p);
  v8bf hi = *(const v8bf*)(p + 16);
  return __builtin_shufflevector(lo, hi, 0,1,2,3,4,5,6,7,8,9,10,11,12,13,14,15);
}

// B (32x16 KxN) from column-major B (= B^T row-major, stride ldt), base at (k0,col0):
//   lane L: g=L>>4, n=L&15 holds column n; elems 0..7 = K(g*8+0..7), 8..15 = K(16+g*8+0..7)
static __device__ __forceinline__ v16bf load_frag_b(const bf16_t* baseT, int ldt, int lane) {
  const int g = lane >> 4, n = lane & 15;
  const bf16_t* p = baseT + (size_t)n * ldt + g * 8;
  v8bf lo = *(const v8bf*)(p);
  v8bf hi = *(const v8bf*)(p + 16);
  return __builtin_shufflevector(lo, hi, 0,1,2,3,4,5,6,7,8,9,10,11,12,13,14,15);
}

// ---------------------------------------------------------------------------
// fp32 -> bf16 bulk convert (vectorized x4)
// ---------------------------------------------------------------------------
__global__ void cvt_f32_bf16(const float* __restrict__ src,
                             unsigned short* __restrict__ dst, int n4) {
  int i = blockIdx.x * blockDim.x + threadIdx.x;
  if (i < n4) {
    const float4 v = ((const float4*)src)[i];
    ushort4 o;
    o.x = f2bf_bits(v.x); o.y = f2bf_bits(v.y);
    o.z = f2bf_bits(v.z); o.w = f2bf_bits(v.w);
    ((ushort4*)dst)[i] = o;
  }
}

// ---------------------------------------------------------------------------
// bf16 GEMM with async double-buffered LDS staging.
// C[M,N] = A[M,K] * B[K,N], B given as B^T row-major (== nn.Linear W).
// Block = 256 thr (8 waves), tile 128x128, K-chunk 64.
// MODE 0: out bf16 -> [b,h,s,hd] (Q,K) | MODE 1: bf16 -> [b,h,hd,s] (V^T)
// MODE 2: out fp32 row-major [M,N]
// ---------------------------------------------------------------------------
template<int MODE>
__global__ __launch_bounds__(256)
void gemm_bf16_k(const bf16_t* __restrict__ A, const bf16_t* __restrict__ Bt,
                 void* __restrict__ out, int M, int N, int K, int lda, int ldb) {
  __shared__ __align__(16) bf16_t As[2][128 * 64];
  __shared__ __align__(16) bf16_t Bs[2][128 * 64];

  const int tid  = threadIdx.x;
  const int lane = tid & 31;
  const int wid  = tid >> 5;
  const int wm = wid & 3, wn = wid >> 2;
  const int rowbase = blockIdx.x * 128;
  const int colbase = blockIdx.y * 128;
  const int rl = wm * 32;   // wave row offset inside tile
  const int cl = wn * 64;   // wave col offset inside tile

  const int seg = tid & 7, r0 = tid >> 3;   // copy mapping: 32 rows x 8 segs per round

  const v8f vz = {};
  v8f acc[2][4];
#pragma unroll
  for (int mt = 0; mt < 2; ++mt)
#pragma unroll
    for (int t = 0; t < 4; ++t) acc[mt][t] = vz;

  auto prefetch = [&](int kc, int buf) {
    const int k0 = kc * 64;
#pragma unroll
    for (int rr = 0; rr < 4; ++rr) {
      const int r = r0 + rr * 32;
      async_copy16(&As[buf][r * 64 + seg * 8], A  + (size_t)(rowbase + r) * lda + k0 + seg * 8);
      async_copy16(&Bs[buf][r * 64 + seg * 8], Bt + (size_t)(colbase + r) * ldb + k0 + seg * 8);
    }
  };

  const int nch = K / 64;
  prefetch(0, 0);
  for (int kc = 0; kc < nch; ++kc) {
    const int cur = kc & 1;
    if (kc + 1 < nch) { prefetch(kc + 1, cur ^ 1); wait_async<8>(); }
    else              { wait_async<0>(); }
    __syncthreads();
#pragma unroll
    for (int ks = 0; ks < 2; ++ks) {
      v16bf a0 = load_frag_a(&As[cur][(rl     ) * 64 + ks * 32], 64, lane);
      v16bf a1 = load_frag_a(&As[cur][(rl + 16) * 64 + ks * 32], 64, lane);
#pragma unroll
      for (int t = 0; t < 4; ++t) {
        v16bf b = load_frag_b(&Bs[cur][(cl + t * 16) * 64 + ks * 32], 64, lane);
        acc[0][t] = WMMA_BF16(a0, b, acc[0][t]);
        acc[1][t] = WMMA_BF16(a1, b, acc[1][t]);
      }
    }
    __syncthreads();
  }

  // C layout: VGPR j -> row (j + 8*(lane>=16)), col = lane%16
  const int g = lane >> 4, cc = lane & 15;
#pragma unroll
  for (int mt = 0; mt < 2; ++mt)
#pragma unroll
    for (int t = 0; t < 4; ++t)
#pragma unroll
      for (int j = 0; j < 8; ++j) {
        const int row = rowbase + rl + mt * 16 + j + g * 8;
        const int col = colbase + cl + t * 16 + cc;
        const float v = acc[mt][t][j];
        if (MODE == 2) {
          ((float*)out)[(size_t)row * N + col] = v;
        } else {
          const int b  = row >> 11, sp = row & 2047;   // SEQ = 2048
          const int h  = col >> 7,  dd = col & 127;    // HD  = 128
          unsigned short* o16 = (unsigned short*)out;
          if (MODE == 0)
            o16[((size_t)(b * 16 + h) * 2048 + sp) * 128 + dd] = f2bf_bits(v);
          else
            o16[((size_t)(b * 16 + h) * 128 + dd) * 2048 + sp] = f2bf_bits(v);
        }
      }
}

// ---------------------------------------------------------------------------
// Flash attention per (b,h): Q[s,128], K[s,128], Vt[128,s] (bf16).
// Block = 8 waves x 16 query rows = 128 queries. K/V chunks (64 keys) are
// async-staged into double-buffered LDS shared by all waves; per-wave online
// softmax; P staged through a wave-private LDS tile for the A-fragment layout.
// ---------------------------------------------------------------------------
__global__ __launch_bounds__(256)
void attn_k(const bf16_t* __restrict__ Q, const bf16_t* __restrict__ Kc,
            const bf16_t* __restrict__ Vt, unsigned short* __restrict__ att) {
  __shared__ __align__(16) bf16_t Ks[2][64 * 128];            // [key][hd]
  __shared__ __align__(16) bf16_t Vs[2][128 * 64];            // [hd][key]
  __shared__ __align__(16) unsigned short plds[8 * 16 * 64];  // P tiles

  const int tid  = threadIdx.x;
  const int lane = tid & 31;
  const int wid  = tid >> 5;
  const int g = lane >> 4, hl = lane & 15;
  const int bh = blockIdx.y;                       // b*16 + h
  const int qrow0 = blockIdx.x * 128 + wid * 16;   // query row within (b,h)

  const bf16_t* Qh = Q  + (size_t)bh * 2048 * 128;
  const bf16_t* Kh = Kc + (size_t)bh * 2048 * 128;
  const bf16_t* Vh = Vt + (size_t)bh * 128 * 2048;

  v16bf qf[4];
#pragma unroll
  for (int kk = 0; kk < 4; ++kk)
    qf[kk] = load_frag_a(Qh + (size_t)qrow0 * 128 + kk * 32, 128, lane);

  const v8f vz = {};
  v8f o[8];
#pragma unroll
  for (int n = 0; n < 8; ++n) o[n] = vz;
  float mrow[8], lrow[8];
#pragma unroll
  for (int j = 0; j < 8; ++j) { mrow[j] = -1e30f; lrow[j] = 0.f; }

  unsigned short* pw = plds + wid * (16 * 64);
  const float SC = 0.08838834764831845f;  // 1/sqrt(128)

  auto prefetch = [&](int kb, int buf) {
#pragma unroll
    for (int rr = 0; rr < 4; ++rr) {
      const int idx = tid + rr * 256;               // 0..1023
      const int kr = idx >> 4, sk = idx & 15;       // Ks: 64 rows x 16 segs
      async_copy16(&Ks[buf][kr * 128 + sk * 8], Kh + (size_t)(kb + kr) * 128 + sk * 8);
      const int dd = idx >> 3, sv = idx & 7;        // Vs: 128 rows x 8 segs
      async_copy16(&Vs[buf][dd * 64 + sv * 8], Vh + (size_t)dd * 2048 + kb + sv * 8);
    }
  };

  const int nch = 2048 / 64;
  prefetch(0, 0);
  for (int c = 0; c < nch; ++c) {
    const int kb = c * 64;
    const int cur = c & 1;
    if (c + 1 < nch) { prefetch(kb + 64, cur ^ 1); wait_async<8>(); }
    else             { wait_async<0>(); }
    __syncthreads();

    // S = Q * K^T for 64 keys: 4 C-tiles x 4 K-steps, K frags from LDS
    v8f s[4];
#pragma unroll
    for (int t = 0; t < 4; ++t) s[t] = vz;
#pragma unroll
    for (int t = 0; t < 4; ++t)
#pragma unroll
      for (int kk = 0; kk < 4; ++kk) {
        v16bf kf = load_frag_b(&Ks[cur][(t * 16) * 128 + kk * 32], 128, lane);
        s[t] = WMMA_BF16(qf[kk], kf, s[t]);
      }

    // Online softmax: row j lives in VGPR j across a 16-lane half
#pragma unroll
    for (int j = 0; j < 8; ++j) {
      float pm = -1e30f;
#pragma unroll
      for (int t = 0; t < 4; ++t) { s[t][j] *= SC; pm = fmaxf(pm, s[t][j]); }
#pragma unroll
      for (int m = 1; m < 16; m <<= 1) pm = fmaxf(pm, __shfl_xor(pm, m, 32));
      const float mn = fmaxf(mrow[j], pm);
      const float alpha = __expf(mrow[j] - mn);
      mrow[j] = mn;
      float rs = 0.f;
#pragma unroll
      for (int t = 0; t < 4; ++t) { float p = __expf(s[t][j] - mn); s[t][j] = p; rs += p; }
#pragma unroll
      for (int m = 1; m < 16; m <<= 1) rs += __shfl_xor(rs, m, 32);
      lrow[j] = lrow[j] * alpha + rs;
#pragma unroll
      for (int n = 0; n < 8; ++n) o[n][j] *= alpha;
#pragma unroll
      for (int t = 0; t < 4; ++t)
        pw[(j + g * 8) * 64 + t * 16 + hl] = f2bf_bits(s[t][j]);
    }

    // O += P * V  (2 K-steps of 32 keys; 8 hd tiles), V frags from LDS
#pragma unroll
    for (int ks = 0; ks < 2; ++ks) {
      v16bf pf = load_frag_a((const bf16_t*)pw + ks * 32, 64, lane);
#pragma unroll
      for (int n = 0; n < 8; ++n) {
        v16bf vf = load_frag_b(&Vs[cur][(n * 16) * 64 + ks * 32], 64, lane);
        o[n] = WMMA_BF16(pf, vf, o[n]);
      }
    }
    __syncthreads();
  }

  // epilogue: normalize and scatter to att[b, s, h*128+dd] (bf16)
  const int b = bh >> 4, h = bh & 15;
#pragma unroll
  for (int j = 0; j < 8; ++j) {
    const float inv = 1.0f / lrow[j];
    const int sp = qrow0 + j + g * 8;
#pragma unroll
    for (int n = 0; n < 8; ++n) {
      const int dd = n * 16 + hl;
      att[((size_t)(b * 2048 + sp)) * 2048 + h * 128 + dd] = f2bf_bits(o[n][j] * inv);
    }
  }
}

// ---------------------------------------------------------------------------
// Host-side orchestration
// ---------------------------------------------------------------------------
extern "C" void kernel_launch(void* const* d_in, const int* in_sizes, int n_in,
                              void* d_out, int out_size, void* d_ws, size_t ws_size,
                              hipStream_t stream) {
  const float* x  = (const float*)d_in[0];
  const float* wq = (const float*)d_in[1];
  const float* wk = (const float*)d_in[2];
  const float* wv = (const float*)d_in[3];
  const float* wo = (const float*)d_in[4];

  const size_t EL_X = 4096ull * 2048;  // 8,388,608
  const size_t EL_W = 2048ull * 2048;  // 4,194,304

  char* ws = (char*)d_ws;
  unsigned short* xb  = (unsigned short*)ws;                // bf16 x; reused for att
  unsigned short* wqb = (unsigned short*)(ws + 2 * EL_X);
  unsigned short* wkb = wqb + EL_W;
  unsigned short* wvb = wkb + EL_W;
  unsigned short* wob = wvb + EL_W;
  unsigned short* qb  = wob + EL_W;                         // [b,h,s,hd]
  unsigned short* kbf = qb + EL_X;                          // [b,h,s,hd]
  unsigned short* vtb = kbf + EL_X;                         // [b,h,hd,s]

  cvt_f32_bf16<<<(int)((EL_X / 4 + 255) / 256), 256, 0, stream>>>(x,  xb,  (int)(EL_X / 4));
  cvt_f32_bf16<<<(int)((EL_W / 4 + 255) / 256), 256, 0, stream>>>(wq, wqb, (int)(EL_W / 4));
  cvt_f32_bf16<<<(int)((EL_W / 4 + 255) / 256), 256, 0, stream>>>(wk, wkb, (int)(EL_W / 4));
  cvt_f32_bf16<<<(int)((EL_W / 4 + 255) / 256), 256, 0, stream>>>(wv, wvb, (int)(EL_W / 4));
  cvt_f32_bf16<<<(int)((EL_W / 4 + 255) / 256), 256, 0, stream>>>(wo, wob, (int)(EL_W / 4));

  dim3 gg(4096 / 128, 2048 / 128);  // 32 x 16
  gemm_bf16_k<0><<<gg, 256, 0, stream>>>((const bf16_t*)xb, (const bf16_t*)wqb, qb,
                                         4096, 2048, 2048, 2048, 2048);
  gemm_bf16_k<0><<<gg, 256, 0, stream>>>((const bf16_t*)xb, (const bf16_t*)wkb, kbf,
                                         4096, 2048, 2048, 2048, 2048);
  gemm_bf16_k<1><<<gg, 256, 0, stream>>>((const bf16_t*)xb, (const bf16_t*)wvb, vtb,
                                         4096, 2048, 2048, 2048, 2048);

  dim3 ga(2048 / 128, 32);  // 16 query-blocks x (b*h)
  attn_k<<<ga, 256, 0, stream>>>((const bf16_t*)qb, (const bf16_t*)kbf,
                                 (const bf16_t*)vtb, xb);

  gemm_bf16_k<2><<<gg, 256, 0, stream>>>((const bf16_t*)xb, (const bf16_t*)wob, d_out,
                                         4096, 2048, 2048, 2048, 2048);
}
